// CAM_Net_16277926052491
// MI455X (gfx1250) — matrix operand order repaired and checked by reading, not compiled, not run on hardware
//
#include <hip/hip_runtime.h>
#include <hip/hip_bf16.h>

// ---------------------------------------------------------------------------
// CAM-Net forward for MI455X (gfx1250, wave32).
// All channel contractions (q/k/v 1x1 convs, expand 3x3 conv) run on
// v_wmma_f32_16x16x32_f16 (f16 inputs, f32 accumulate).  Attention/softmax in
// f32 VALU.  Intermediates live in d_ws (~43 MB, L2-resident on 192MB L2).
// The expand-conv GEMM stages its B-operand tile (att_scale neighborhood,
// 32ch x 4rows x 32cols = 16KB) into LDS with the Tensor Data Mover
// (TENSOR_LOAD_TO_LDS + s_wait_tensorcnt), one DMA per workgroup.
//
// WMMA operand packing used here (16-bit, wave32; contiguous k, not the ISA table order):
//   A 16x32 : lane L holds row M=L%16 ; elements e:
//             K = (L>=16 ? 16 : 0) + e
//   B 32x16 : lane L holds col N=L%16 ; element e: K = (L>=16 ? 16 : 0) + e
//   C/D 16x16 f32: VGPR r: lanes 0-15 -> (M=r, N=lane), lanes 16-31 -> (M=8+r)
// ---------------------------------------------------------------------------

typedef __attribute__((ext_vector_type(16))) _Float16     v16h;
typedef __attribute__((ext_vector_type(8)))  float        v8f;
typedef __attribute__((ext_vector_type(4)))  unsigned int u32x4;
typedef __attribute__((ext_vector_type(8)))  int          i32x8;
typedef __attribute__((ext_vector_type(4)))  int          i32x4;

#define WMMA_F16(A, B, C) \
  __builtin_amdgcn_wmma_f32_16x16x32_f16(false, (A), false, (B), (short)0, (C), false, false)

static constexpr int H_IMG = 256, W_IMG = 256;
static constexpr int NPIX  = H_IMG * W_IMG;          // 65536
static constexpr int HP    = 262, WP = 262;          // padded by 3 each side
static constexpr int NPP   = HP * WP;                // 68644
static constexpr int C_ATT = 32;
static constexpr int C_FC  = 512;
static constexpr int OUTHW = 32;                     // stride-8 trunk output
static constexpr int NPOUT = OUTHW * OUTHW;          // 1024

// workspace offsets (in floats)
static constexpr size_t OFF_X1P   = 0;                          // 32*NPP padded x1
static constexpr size_t OFF_K     = OFF_X1P   + (size_t)C_ATT * NPP;
static constexpr size_t OFF_V     = OFF_K     + (size_t)C_ATT * NPP;
static constexpr size_t OFF_Q     = OFF_V     + (size_t)C_ATT * NPP;   // 32*NPIX
static constexpr size_t OFF_ATT   = OFF_Q     + (size_t)C_ATT * NPIX;  // 32*NPIX
static constexpr size_t OFF_SCALE = OFF_ATT   + (size_t)C_ATT * NPIX;  // 32*1024
static constexpr size_t OFF_POOL  = OFF_SCALE + (size_t)C_ATT * NPOUT; // 512
static constexpr size_t OFF_REL   = OFF_POOL  + C_FC;                  // 14

// ---------------------------------------------------------------------------
// 0) zero pooled accumulator; reduce rel_h/rel_w over channel dim
// ---------------------------------------------------------------------------
__global__ void k_init(const float* __restrict__ rel_h, const float* __restrict__ rel_w,
                       float* __restrict__ pooled, float* __restrict__ rel) {
  int t = threadIdx.x;
  if (t < C_FC) pooled[t] = 0.f;
  if (t < 7) { float s = 0.f; for (int c = 0; c < 16; ++c) s += rel_h[c * 7 + t]; rel[t] = s; }
  if (t >= 16 && t < 23) {
    int j = t - 16; float s = 0.f; for (int c = 0; c < 16; ++c) s += rel_w[c * 7 + j];
    rel[7 + j] = s;
  }
}

// ---------------------------------------------------------------------------
// 1) conv1: 6->32, 3x3, pad 1, on 256x256; write into zero-padded [32,262,262]
// ---------------------------------------------------------------------------
__global__ void k_conv1(const float* __restrict__ x, const float* __restrict__ w,
                        const float* __restrict__ b, float* __restrict__ x1p) {
  int t = blockIdx.x * blockDim.x + threadIdx.x;
  if (t >= C_ATT * NPP) return;
  int c = t / NPP, r = t % NPP;
  int py = r / WP, px = r % WP;
  int y = py - 3, x0 = px - 3;
  if (y < 0 || y >= H_IMG || x0 < 0 || x0 >= W_IMG) { x1p[t] = 0.f; return; }
  float acc = b[c];
  #pragma unroll
  for (int ic = 0; ic < 6; ++ic)
    #pragma unroll
    for (int ky = 0; ky < 3; ++ky) {
      int yy = y - 1 + ky;
      if (yy < 0 || yy >= H_IMG) continue;
      #pragma unroll
      for (int kx = 0; kx < 3; ++kx) {
        int xx = x0 - 1 + kx;
        if (xx < 0 || xx >= W_IMG) continue;
        acc += x[ic * NPIX + yy * W_IMG + xx] * w[((c * 6 + ic) * 3 + ky) * 3 + kx];
      }
    }
  x1p[t] = acc;
}

// ---------------------------------------------------------------------------
// 2) k & v 1x1 convs over the padded grid: C[32,NPP] = W[32,32] * X1P[32,NPP]
//    One wave per 16-pixel group; 2 M-tiles x {k,v} = 4 WMMAs per group.
// ---------------------------------------------------------------------------
__global__ void k_kv1x1(const float* __restrict__ x1p,
                        const float* __restrict__ kw, const float* __restrict__ kbias,
                        const float* __restrict__ vw, const float* __restrict__ vbias,
                        float* __restrict__ kout, float* __restrict__ vout) {
  const int nGroups = (NPP + 15) >> 4;
  int gwave  = (blockIdx.x * blockDim.x + threadIdx.x) >> 5;
  int nWaves = (gridDim.x * blockDim.x) >> 5;
  int lane = threadIdx.x & 31;
  int mrow = lane & 15;
  int aKhi = (lane & 16) ? 8 : 0;
  int bKhi = (lane & 16) ? 16 : 0;
  v16h aK0, aK1, aV0, aV1;
  #pragma unroll
  for (int e = 0; e < 16; ++e) {
    int kk = e + bKhi;
    aK0[e] = (_Float16)kw[mrow * 32 + kk];
    aK1[e] = (_Float16)kw[(16 + mrow) * 32 + kk];
    aV0[e] = (_Float16)vw[mrow * 32 + kk];
    aV1[e] = (_Float16)vw[(16 + mrow) * 32 + kk];
  }
  for (int g = gwave; g < nGroups; g += nWaves) {
    int p = g * 16 + mrow;
    bool ok = p < NPP;
    int pc = ok ? p : 0;
    v16h b;
    #pragma unroll
    for (int e = 0; e < 16; ++e) {
      float v = x1p[(size_t)(bKhi + e) * NPP + pc];
      b[e] = (_Float16)(ok ? v : 0.f);
    }
    v8f ck0 = {}, ck1 = {}, cv0 = {}, cv1 = {};
    ck0 = WMMA_F16(aK0, b, ck0);
    ck1 = WMMA_F16(aK1, b, ck1);
    cv0 = WMMA_F16(aV0, b, cv0);
    cv1 = WMMA_F16(aV1, b, cv1);
    if (ok) {
      #pragma unroll
      for (int r = 0; r < 8; ++r) {
        int M = (lane < 16) ? r : (8 + r);
        kout[(size_t)M * NPP + p]        = ck0[r] + kbias[M];
        kout[(size_t)(16 + M) * NPP + p] = ck1[r] + kbias[16 + M];
        vout[(size_t)M * NPP + p]        = cv0[r] + vbias[M];
        vout[(size_t)(16 + M) * NPP + p] = cv1[r] + vbias[16 + M];
      }
    }
  }
}

// ---------------------------------------------------------------------------
// 3) q 1x1 conv on the 256x256 interior (reads x1p interior)
// ---------------------------------------------------------------------------
__global__ void k_q1x1(const float* __restrict__ x1p,
                       const float* __restrict__ qw, const float* __restrict__ qbias,
                       float* __restrict__ qout) {
  const int nGroups = NPIX >> 4;  // 4096 exact
  int gwave  = (blockIdx.x * blockDim.x + threadIdx.x) >> 5;
  int nWaves = (gridDim.x * blockDim.x) >> 5;
  int lane = threadIdx.x & 31;
  int mrow = lane & 15;
  int aKhi = (lane & 16) ? 8 : 0;
  int bKhi = (lane & 16) ? 16 : 0;
  v16h aQ0, aQ1;
  #pragma unroll
  for (int e = 0; e < 16; ++e) {
    int kk = e + bKhi;
    aQ0[e] = (_Float16)qw[mrow * 32 + kk];
    aQ1[e] = (_Float16)qw[(16 + mrow) * 32 + kk];
  }
  for (int g = gwave; g < nGroups; g += nWaves) {
    int p = g * 16 + mrow;
    int y = p >> 8, xx = p & 255;
    size_t ib = (size_t)(y + 3) * WP + (xx + 3);
    v16h b;
    #pragma unroll
    for (int e = 0; e < 16; ++e)
      b[e] = (_Float16)x1p[(size_t)(bKhi + e) * NPP + ib];
    v8f c0 = {}, c1 = {};
    c0 = WMMA_F16(aQ0, b, c0);
    c1 = WMMA_F16(aQ1, b, c1);
    #pragma unroll
    for (int r = 0; r < 8; ++r) {
      int M = (lane < 16) ? r : (8 + r);
      qout[(size_t)M * NPIX + p]        = c0[r] + qbias[M];
      qout[(size_t)(16 + M) * NPIX + p] = c1[r] + qbias[16 + M];
    }
  }
}

// ---------------------------------------------------------------------------
// 4) local attention (after the reshape-bijection simplification):
//    one thread per (c, h, wgroup-of-4).
// ---------------------------------------------------------------------------
__global__ void k_attn(const float* __restrict__ qb, const float* __restrict__ kb,
                       const float* __restrict__ vb, const float* __restrict__ rel,
                       float* __restrict__ att) {
  int t = blockIdx.x * blockDim.x + threadIdx.x;
  if (t >= C_ATT * H_IMG * 64) return;
  int wg = t & 63, h = (t >> 6) & 255, c = t >> 14;
  int wbase = wg << 2;
  const float* qp = qb + (size_t)c * NPIX + h * W_IMG + wbase;
  float q0 = qp[0], q1 = qp[1], q2 = qp[2], q3 = qp[3];
  float qs = q0 + q1 + q2 + q3;
  float rh[7], rw[7];
  #pragma unroll
  for (int i = 0; i < 7; ++i) { rh[i] = rel[i]; rw[i] = rel[7 + i]; }
  const float* kc = kb + (size_t)c * NPP;
  const float* vc = vb + (size_t)c * NPP;
  float l[49];
  float m = -1e30f;
  #pragma unroll
  for (int i = 0; i < 7; ++i) {
    const float* kr = kc + (size_t)(h + i) * WP + wbase;
    __builtin_prefetch(vc + (size_t)(h + i) * WP + wbase, 0, 1);  // global_prefetch_b8
    float kvv[10];
    #pragma unroll
    for (int u = 0; u < 10; ++u) kvv[u] = kr[u];
    #pragma unroll
    for (int j = 0; j < 7; ++j) {
      float lv = q0 * kvv[j] + q1 * kvv[j + 1] + q2 * kvv[j + 2] + q3 * kvv[j + 3]
               + qs * (rh[i] + rw[j]);
      l[i * 7 + j] = lv;
      m = fmaxf(m, lv);
    }
  }
  float s = 0.f;
  #pragma unroll
  for (int z = 0; z < 49; ++z) { float e = __expf(l[z] - m); l[z] = e; s += e; }
  float inv = 1.f / s;
  float o0 = 0, o1 = 0, o2 = 0, o3 = 0;
  #pragma unroll
  for (int i = 0; i < 7; ++i) {
    const float* vr = vc + (size_t)(h + i) * WP + wbase;
    float vv[10];
    #pragma unroll
    for (int u = 0; u < 10; ++u) vv[u] = vr[u];
    #pragma unroll
    for (int j = 0; j < 7; ++j) {
      float wgt = l[i * 7 + j] * inv;
      o0 += wgt * vv[j]; o1 += wgt * vv[j + 1]; o2 += wgt * vv[j + 2]; o3 += wgt * vv[j + 3];
    }
  }
  float* op = att + (size_t)c * NPIX + h * W_IMG + wbase;
  op[0] = o0; op[1] = o1; op[2] = o2; op[3] = o3;
}

// ---------------------------------------------------------------------------
// 5) bilinear 8x downsample (align_corners=False -> 2x2 average at +3,+4)
// ---------------------------------------------------------------------------
__global__ void k_down(const float* __restrict__ att, float* __restrict__ scale) {
  int t = blockIdx.x * blockDim.x + threadIdx.x;
  if (t >= C_ATT * NPOUT) return;
  int c = t >> 10, r = t & 1023;
  int oy = r >> 5, ox = r & 31;
  int y0 = 8 * oy + 3, x0 = 8 * ox + 3;
  const float* a = att + (size_t)c * NPIX;
  scale[t] = 0.25f * (a[y0 * W_IMG + x0] + a[y0 * W_IMG + x0 + 1] +
                      a[(y0 + 1) * W_IMG + x0] + a[(y0 + 1) * W_IMG + x0 + 1]);
}

// ---------------------------------------------------------------------------
// 6) backbone stride-8 conv, reduced straight to pooled[] mean (+biases)
// ---------------------------------------------------------------------------
__global__ void k_backbone_pool(const float* __restrict__ x, const float* __restrict__ w,
                                const float* __restrict__ b, const float* __restrict__ eb,
                                float* __restrict__ pooled) {
  int oc = blockIdx.x, tid = threadIdx.x;
  float s = 0.f;
  for (int p = tid; p < NPOUT; p += blockDim.x) {
    int oy = p >> 5, ox = p & 31;
    float acc = 0.f;
    #pragma unroll
    for (int ic = 0; ic < 6; ++ic)
      #pragma unroll
      for (int ky = 0; ky < 3; ++ky) {
        int yy = 8 * oy - 1 + ky;
        if (yy < 0 || yy >= H_IMG) continue;
        #pragma unroll
        for (int kx = 0; kx < 3; ++kx) {
          int xx = 8 * ox - 1 + kx;
          if (xx < 0 || xx >= W_IMG) continue;
          acc += x[ic * NPIX + yy * W_IMG + xx] * w[((oc * 6 + ic) * 3 + ky) * 3 + kx];
        }
      }
    s += acc;
  }
  __shared__ float red[256];
  red[tid] = s;
  __syncthreads();
  for (int stp = 128; stp > 0; stp >>= 1) {
    if (tid < stp) red[tid] += red[tid + stp];
    __syncthreads();
  }
  if (tid == 0) pooled[oc] += b[oc] + eb[oc] + red[0] * (1.f / (float)NPOUT);
}

// ---------------------------------------------------------------------------
// 7) expand conv as im2col GEMM [512x288]x[288x1024] via WMMA.
//    B-operand neighborhood (32ch x 4rows x 32cols, 16KB) is DMA-staged into
//    LDS once per workgroup by the Tensor Data Mover (TENSOR_LOAD_TO_LDS,
//    TENSORcnt-tracked); pooled mean folded into the epilogue via wave
//    shuffles + atomics (att_enhance never materialized).
// ---------------------------------------------------------------------------
__global__ void k_expand_pool(const float* __restrict__ scale, const float* __restrict__ w,
                              float* __restrict__ pooled) {
  __shared__ float tile[C_ATT * 4 * OUTHW];   // [ic][row][x], 16 KB
  int lane = threadIdx.x & 31;
  int wv   = threadIdx.x >> 5;
  int mt = blockIdx.x;            // 0..31  (16 out-channels each)
  int nt = blockIdx.y * 4 + wv;   // 0..63  (16 pixels each)
  int yb = blockIdx.y * 2;        // first output row this block touches
  int ys = yb - 1; if (ys < 0) ys = 0; if (ys > OUTHW - 4) ys = OUTHW - 4;  // staged rows [ys, ys+4)

  // --- TDM: stage att_scale[0..31][ys..ys+3][0..31] into LDS ---------------
  if (wv == 0) {
    unsigned lds_a = (unsigned)(unsigned long long)(&tile[0]);   // low 32b = LDS offset
    unsigned long long ga = (unsigned long long)(const void*)(scale + (size_t)ys * OUTHW);
    // D# group 0: count=1 | lds_addr | global_addr | type=2 ("image")
    u32x4 g0 = { 1u, lds_a, (unsigned)ga, (unsigned)(ga >> 32) | 0x80000000u };
    // D# group 1: data_size=4B; tensor_dim0=32, tensor_dim1=4;
    //             tile_dim0=32, tile_dim1=4, tile_dim2=32;
    //             dim0_stride=32 (row pitch), dim1_stride=1024 (channel pitch)
    i32x8 g1 = { 0x00020000,
                 32 << 16,            // tensor_dim0 low16 @ bits[63:48]
                 4 << 16,             // tensor_dim1 low16 @ bits[95:80]
                 32 << 16,            // tile_dim0 @ bits[127:112]
                 (32 << 16) | 4,      // tile_dim2 | tile_dim1
                 32,                  // tensor_dim0_stride low32
                 1024 << 16,          // tensor_dim1_stride low16 @ bits[223:208]
                 0 };
    i32x4 g2 = { 32, 0, 0, 0 };        // tensor_dim2 = 32 (channels)
    i32x4 g3 = { 0, 0, 0, 0 };
#if __clang_major__ >= 23
    i32x8 g4 = { 0, 0, 0, 0, 0, 0, 0, 0 };
    __builtin_amdgcn_tensor_load_to_lds(g0, g1, g2, g3, g4, 0);
#else
    __builtin_amdgcn_tensor_load_to_lds(g0, g1, g2, g3, 0);
#endif
    __builtin_amdgcn_s_wait_tensorcnt(0);
  }
  __syncthreads();

  int mrow = lane & 15;
  int p = nt * 16 + mrow;
  int oy = p >> 5, ox = p & 31;
  int aKhi = (lane & 16) ? 8 : 0;
  int bKhi = (lane & 16) ? 16 : 0;
  v8f acc = {};
  for (int kt = 0; kt < 9; ++kt) {
    v16h a, bv;
    #pragma unroll
    for (int e = 0; e < 16; ++e) {
      int kA = kt * 32 + bKhi + e;
      a[e] = (_Float16)w[(size_t)(mt * 16 + mrow) * 288 + kA];
      int kB = kt * 32 + bKhi + e;
      int ic = kB / 9, rr = kB % 9, ky = rr / 3, kx = rr % 3;
      int y = oy - 1 + ky, xx = ox - 1 + kx;
      float v = 0.f;
      if (y >= 0 && y < OUTHW && xx >= 0 && xx < OUTHW)
        v = tile[ic * (4 * OUTHW) + (y - ys) * OUTHW + xx];
      bv[e] = (_Float16)v;
    }
    acc = WMMA_F16(a, bv, acc);
  }
  #pragma unroll
  for (int r = 0; r < 8; ++r) {
    float s = acc[r];
    s += __shfl_xor(s, 1); s += __shfl_xor(s, 2);
    s += __shfl_xor(s, 4); s += __shfl_xor(s, 8);
    if ((lane & 15) == 0) {
      int M = (lane < 16) ? r : (8 + r);
      atomicAdd(&pooled[mt * 16 + M], s * (1.f / (float)NPOUT));
    }
  }
}

// ---------------------------------------------------------------------------
// 8) fc: out[20] = pooled[512] @ fc_w.T + fc_b
// ---------------------------------------------------------------------------
__global__ void k_fc(const float* __restrict__ pooled, const float* __restrict__ fw,
                     const float* __restrict__ fb, float* __restrict__ out) {
  int c = threadIdx.x;
  if (c >= 20) return;
  float s = fb[c];
  for (int i = 0; i < C_FC; ++i) s += pooled[i] * fw[c * C_FC + i];
  out[c] = s;
}

// ---------------------------------------------------------------------------
extern "C" void kernel_launch(void* const* d_in, const int* in_sizes, int n_in,
                              void* d_out, int out_size, void* d_ws, size_t ws_size,
                              hipStream_t stream) {
  (void)in_sizes; (void)n_in; (void)out_size; (void)ws_size;
  const float* x          = (const float*)d_in[0];
  const float* backbone_w = (const float*)d_in[2];
  const float* backbone_b = (const float*)d_in[3];
  const float* conv1_w    = (const float*)d_in[4];
  const float* conv1_b    = (const float*)d_in[5];
  const float* k_w        = (const float*)d_in[6];
  const float* k_b        = (const float*)d_in[7];
  const float* q_w        = (const float*)d_in[8];
  const float* q_b        = (const float*)d_in[9];
  const float* v_w        = (const float*)d_in[10];
  const float* v_b        = (const float*)d_in[11];
  const float* rel_h      = (const float*)d_in[12];
  const float* rel_w      = (const float*)d_in[13];
  const float* expand_w   = (const float*)d_in[14];
  const float* expand_b   = (const float*)d_in[15];
  const float* fc_w       = (const float*)d_in[16];
  const float* fc_b       = (const float*)d_in[17];
  float* out = (float*)d_out;

  float* ws     = (float*)d_ws;
  float* x1p    = ws + OFF_X1P;
  float* kbuf   = ws + OFF_K;
  float* vbuf   = ws + OFF_V;
  float* qbuf   = ws + OFF_Q;
  float* attbuf = ws + OFF_ATT;
  float* scl    = ws + OFF_SCALE;
  float* pooled = ws + OFF_POOL;
  float* rel    = ws + OFF_REL;

  k_init<<<1, 512, 0, stream>>>(rel_h, rel_w, pooled, rel);
  k_conv1<<<(C_ATT * NPP + 255) / 256, 256, 0, stream>>>(x, conv1_w, conv1_b, x1p);
  k_kv1x1<<<512, 256, 0, stream>>>(x1p, k_w, k_b, v_w, v_b, kbuf, vbuf);
  k_q1x1<<<512, 256, 0, stream>>>(x1p, q_w, q_b, qbuf);
  k_attn<<<(C_ATT * H_IMG * 64) / 256, 256, 0, stream>>>(qbuf, kbuf, vbuf, rel, attbuf);
  k_down<<<(C_ATT * NPOUT + 255) / 256, 256, 0, stream>>>(attbuf, scl);
  k_backbone_pool<<<C_FC, 256, 0, stream>>>(x, backbone_w, backbone_b, expand_b, pooled);
  k_expand_pool<<<dim3(32, 16), 128, 0, stream>>>(scl, expand_w, pooled);
  k_fc<<<1, 32, 0, stream>>>(pooled, fc_w, fc_b, out);
}
